// FoatAttention_79680233275829
// MI455X (gfx1250) — matrix operand order repaired and checked by reading, not compiled, not run on hardware
//
#include <hip/hip_runtime.h>
#include <hip/hip_bf16.h>
#include <math.h>

// ---------------------------------------------------------------------------
// FoatAttention block for MI455X (gfx1250), wave32 + WMMA bf16 + TDM.
// All dense algebra (qkv/proj/fc1/fc2, truncated-DFT fwd/inv transforms,
// spectral mode mixing, factorized attention kv/fa) runs through
// v_wmma_f32_16x16x32_bf16 with f32 accumulation. A tiles are staged into
// LDS by the Tensor Data Mover (tensor_load_to_lds / TENSORcnt); B tiles are
// staged transposed so B fragments are contiguous ds_load_b128 reads.
//
// GEMM contract (all call sites obey):
//   K % 32 == 0, lda % 8 == 0, ldb % 8 == 0, base pointers 16B-aligned,
//   operands readable up to the 128-padded tile boundary (allocator adds
//   4 KB slack per buffer) -> staging carries NO bounds checks.
//   Stores are exact (guarded). Workspace requirement: ~1.1 GB.
// ---------------------------------------------------------------------------

typedef __attribute__((ext_vector_type(16))) __bf16 bh16;
typedef __attribute__((ext_vector_type(8)))  __bf16 bh8;
typedef __attribute__((ext_vector_type(8)))  float  f32x8;
typedef unsigned int u32x4 __attribute__((ext_vector_type(4)));
typedef int          i32x8 __attribute__((ext_vector_type(8)));
typedef int          i32x4 __attribute__((ext_vector_type(4)));

#if __has_builtin(__builtin_amdgcn_tensor_load_to_lds) && \
    __has_builtin(__builtin_amdgcn_s_wait_tensorcnt)
#define HAVE_TDM 1
#else
#define HAVE_TDM 0
#endif

#if HAVE_TDM
// 2D-tile TDM load: tile_rows x tile_k bf16 elements, row stride stride_elems,
// global -> LDS at lds_byte_addr. D# bit layout per CDNA5 ISA §8.3/8.4.
// This toolchain exposes the 6-arg builtin (groups 2/3 + extra group, zeroed
// for <=2D tiles) per the amdgpu-toolchain signature.
static __device__ __forceinline__
void tdm_load_tile_2d(unsigned lds_byte_addr, const void* gptr,
                      unsigned tile_k, unsigned tile_rows,
                      unsigned long long stride_elems)
{
    unsigned long long ga = (unsigned long long)(size_t)gptr;   // byte address
    u32x4 g0;
    g0.x = 1u;                                       // count=1 (valid), user mode
    g0.y = lds_byte_addr;                            // lds_addr [63:32]
    g0.z = (unsigned)(ga & 0xffffffffu);             // global_addr [95:64]
    g0.w = (unsigned)((ga >> 32) & 0x01ffffffu)      // global_addr [120:96]
         | (2u << 30);                               // type=2 ("image")
    const unsigned td0 = 0x40000000u, td1 = 0x40000000u;  // huge dims: no OOB clip
    i32x8 g1;
    g1[0] = (int)(1u << 16);                         // data_size=1 (2 bytes)
    g1[1] = (int)((td0 & 0xffffu) << 16);            // tensor_dim0 [15:0]
    g1[2] = (int)((td0 >> 16) | ((td1 & 0xffffu) << 16));  // dim0 hi | dim1 lo
    g1[3] = (int)((td1 >> 16) | (tile_k << 16));     // dim1 hi | tile_dim0
    g1[4] = (int)(tile_rows & 0xffffu);              // tile_dim1 (tile_dim2=0)
    g1[5] = (int)(stride_elems & 0xffffffffull);     // tensor_dim0_stride [31:0]
    g1[6] = (int)((stride_elems >> 32) & 0xffffull); // tensor_dim0_stride [47:32]
    g1[7] = 0;                                       // tensor_dim1_stride unused
    __builtin_amdgcn_tensor_load_to_lds(g0, g1, (i32x4)0, (i32x4)0, (i32x8)0, 0);
}
#endif

// ======================= generic bf16 GEMM (WMMA) ==========================
// C[M,N](f32) = A[M,K](bf16,rm) * B[K,N](bf16,rm) (+bias)(+resid)
// 256 thr = 8 waves; block tile 128x128; wave tile 32x64 (2x4 wmma).
__global__ __launch_bounds__(256)
void k_gemm(const __bf16* __restrict__ Ag, const __bf16* __restrict__ Bg,
            float* __restrict__ Cg, int M, int N, int K,
            int lda, int ldb, int ldc,
            long long sA, long long sB, long long sC,
            const float* __restrict__ bias, const float* __restrict__ resid)
{
    const int bz = blockIdx.z;
    const __bf16* A = Ag + sA * bz;
    const __bf16* B = Bg + sB * bz;
    float*       C  = Cg + sC * bz;
    const float* R  = resid ? (resid + sC * bz) : nullptr;

    __shared__ __align__(16) __bf16 As[128][32];   // row-major A tile
    __shared__ __align__(16) __bf16 Bs[128][32];   // TRANSPOSED B tile: Bs[n][k]

    const int tid  = threadIdx.x;
    const int lane = tid & 31;
    const int wave = tid >> 5;
    const int wm   = wave & 3;       // 4 row groups of 32
    const int wn   = wave >> 2;      // 2 col groups of 64
    const int m0   = blockIdx.y * 128;
    const int n0   = blockIdx.x * 128;

    f32x8 acc[2][4] = {};
    const int row16 = lane & 15;
    const int kb8   = (lane >> 4) << 3;   // A frag K base (ISA 16-bit A layout)
    const int kb16  = (lane >> 4) << 4;   // unused; B frags use kb8 like A

    for (int k0 = 0; k0 < K; k0 += 32) {
#if HAVE_TDM
        // async DMA of the 128x32 A tile straight into LDS (no VGPR round trip);
        // overlaps with the manual transposed B staging below.
        if (wave == 0)
            tdm_load_tile_2d((unsigned)(size_t)&As[0][0],
                             A + (long long)m0 * lda + k0,
                             32u, 128u, (unsigned long long)lda);
#else
        #pragma unroll
        for (int e = 0; e < 2; ++e) {
            int id = e * 256 + tid;
            int r = id >> 2, c8 = id & 3;
            bh8 v = *(const bh8*)(A + (long long)(m0 + r) * lda + k0 + 8 * c8);
            *(bh8*)(&As[r][8 * c8]) = v;
        }
#endif
        // stage B transposed: 32k x 128n = 512 chunks of 8, 2 per thread
        #pragma unroll
        for (int e = 0; e < 2; ++e) {
            int id = e * 256 + tid;
            int kk = id >> 4, c8 = id & 15;
            bh8 v = *(const bh8*)(B + (long long)(k0 + kk) * ldb + n0 + 8 * c8);
            #pragma unroll
            for (int j = 0; j < 8; ++j) Bs[8 * c8 + j][kk] = v[j];
        }
        if (k0 + 32 < K)   // global_prefetch_b8 hint on the B stream
            __builtin_prefetch(B + (long long)(k0 + 32 + (tid >> 4)) * ldb + n0, 0, 1);
#if HAVE_TDM
        if (wave == 0) __builtin_amdgcn_s_wait_tensorcnt((short)0);
#endif
        __syncthreads();

        bh16 af[2], bf[4];
        #pragma unroll
        for (int tm = 0; tm < 2; ++tm) {
            int ar = wm * 32 + tm * 16 + row16;
            bh8 lo = *(const bh8*)(&As[ar][kb8]);
            bh8 hi = *(const bh8*)(&As[ar][16 + kb8]);
            af[tm] = __builtin_shufflevector(lo, hi, 0,1,2,3,4,5,6,7,8,9,10,11,12,13,14,15);
        }
        #pragma unroll
        for (int tn = 0; tn < 4; ++tn) {
            int bc = wn * 64 + tn * 16 + row16;
            bh8 lo = *(const bh8*)(&Bs[bc][kb8]);
            bh8 hi = *(const bh8*)(&Bs[bc][16 + kb8]);
            bf[tn] = __builtin_shufflevector(lo, hi, 0,1,2,3,4,5,6,7,8,9,10,11,12,13,14,15);
        }
        #pragma unroll
        for (int tm = 0; tm < 2; ++tm)
            #pragma unroll
            for (int tn = 0; tn < 4; ++tn)
                acc[tm][tn] = __builtin_amdgcn_wmma_f32_16x16x32_bf16(
                    false, af[tm], false, bf[tn], (short)0, acc[tm][tn], false, false);
        __syncthreads();
    }

    // epilogue: C layout -> M = v + 8*(lane>=16), N = lane&15
    const int mib = (lane >> 4) << 3;
    const int ni  = lane & 15;
    #pragma unroll
    for (int tm = 0; tm < 2; ++tm)
        #pragma unroll
        for (int tn = 0; tn < 4; ++tn) {
            int gc = n0 + wn * 64 + tn * 16 + ni;
            if (gc >= N) continue;
            #pragma unroll
            for (int v = 0; v < 8; ++v) {
                int gr = m0 + wm * 32 + tm * 16 + mib + v;
                if (gr < M) {
                    float val = acc[tm][tn][v];
                    if (bias)  val += bias[gc];
                    if (resid) val += R[(long long)gr * ldc + gc];
                    C[(long long)gr * ldc + gc] = val;
                }
            }
        }
}

// ================= spectral mode mixing (complex C×C per mode) =============
// For mode m: A = [xr|xi]  (16 x 1536, from XT[m]),
//             B(k,j) built on the fly from w1r/w1i (m<28) or w2r/w2i:
//   j<768 (real out):  k<768 -> wr[i][o],  k>=768 -> -wi[i][o]
//   j>=768 (imag out): k<768 -> wi[i][o],  k>=768 ->  wr[i][o]
// Output scattered straight into the inverse-transform A operand:
//   A3[(b*768+o)][2m+p]  (p = j>=768)
__global__ __launch_bounds__(256)
void k_modemix(const __bf16* __restrict__ XT,      // [56][16][1536]
               const float* __restrict__ w1r, const float* __restrict__ w1i,
               const float* __restrict__ w2r, const float* __restrict__ w2i,
               __bf16* __restrict__ A3)             // [12288][128]
{
    const int m  = blockIdx.y;
    const int n0 = blockIdx.x * 128;
    const float* wr = (m < 28) ? w1r : w2r;
    const float* wi = (m < 28) ? w1i : w2i;
    const int mx = (m < 28) ? (m >> 2) : ((m - 28) >> 2);
    const int woff = mx * 4 + (m & 3);
    const __bf16* Arow = XT + (size_t)m * 16 * 1536;

    __shared__ __align__(16) __bf16 As[16][32];
    __shared__ __align__(16) __bf16 Bs[128][32];   // transposed

    const int tid = threadIdx.x, lane = tid & 31, wave = tid >> 5;
    f32x8 acc = {};
    const int row16 = lane & 15, kb8 = (lane >> 4) << 3, kb16 = (lane >> 4) << 4;

    for (int k0 = 0; k0 < 1536; k0 += 32) {
#if HAVE_TDM
        if (wave == 0)
            tdm_load_tile_2d((unsigned)(size_t)&As[0][0], Arow + k0,
                             32u, 16u, 1536ull);
#else
        if (tid < 64) {                 // 16x32 = 64 chunks of 8
            int r = tid >> 2, c8 = tid & 3;
            bh8 v = *(const bh8*)(Arow + (size_t)r * 1536 + k0 + 8 * c8);
            *(bh8*)(&As[r][8 * c8]) = v;
        }
#endif
        #pragma unroll
        for (int e = 0; e < 16; ++e) {  // strided f32 weight gather (branchless)
            int idx = e * 256 + tid;
            int col = idx & 127, kk = idx >> 7;
            int gk = k0 + kk, gj = n0 + col;
            int i = (gk < 768) ? gk : gk - 768;
            int o = (gj < 768) ? gj : gj - 768;
            size_t widx = ((size_t)i * 768 + o) * 28 + woff;
            float v;
            if (gj < 768) v = (gk < 768) ? wr[widx] : -wi[widx];
            else          v = (gk < 768) ? wi[widx] :  wr[widx];
            Bs[col][kk] = (__bf16)v;
        }
#if HAVE_TDM
        if (wave == 0) __builtin_amdgcn_s_wait_tensorcnt((short)0);
#endif
        __syncthreads();
        bh16 af, bv;
        {
            bh8 lo = *(const bh8*)(&As[row16][kb8]);
            bh8 hi = *(const bh8*)(&As[row16][16 + kb8]);
            af = __builtin_shufflevector(lo, hi, 0,1,2,3,4,5,6,7,8,9,10,11,12,13,14,15);
        }
        {
            int bc = wave * 16 + row16;
            bh8 lo = *(const bh8*)(&Bs[bc][kb8]);
            bh8 hi = *(const bh8*)(&Bs[bc][16 + kb8]);
            bv = __builtin_shufflevector(lo, hi, 0,1,2,3,4,5,6,7,8,9,10,11,12,13,14,15);
        }
        acc = __builtin_amdgcn_wmma_f32_16x16x32_bf16(false, af, false, bv,
                                                      (short)0, acc, false, false);
        __syncthreads();
    }
    const int mib = (lane >> 4) << 3, ni = lane & 15;
    #pragma unroll
    for (int v = 0; v < 8; ++v) {
        int b  = mib + v;                // batch row (0..15)
        int gj = n0 + wave * 16 + ni;
        int o  = (gj < 768) ? gj : gj - 768;
        int p  = (gj >= 768);
        A3[((size_t)b * 768 + o) * 128 + 2 * m + p] = (__bf16)acc[v];
    }
}

// ===================== truncated-DFT basis tables ==========================
// Forward: Ff[n][2m]=cosθ, Ff[n][2m+1]=-sinθ  (θ=2π(kx·h+ky·w)/32)
// Inverse (irfft2 with Hermitian ky doubling): Bi[2m][n]=(c/1024)cosθ,
//   Bi[2m+1][n]=-(c/1024)sinθ, c = ky? 2:1.  Rows/cols 112..127 zero-padded.
__global__ void k_basis(__bf16* __restrict__ Ff, __bf16* __restrict__ Bi)
{
    int n = blockIdx.x;          // pixel 0..1023
    int col = threadIdx.x;       // 0..127
    int m = col >> 1, p = col & 1;
    float fwd = 0.f, inv = 0.f;
    if (m < 56) {
        int kx = (m < 28) ? (m >> 2) : (25 + ((m - 28) >> 2));
        int ky = m & 3;
        int h = n >> 5, w = n & 31;
        float th = 6.283185307179586f * (float)((kx * h + ky * w) & 31) / 32.f;
        float c = cosf(th), s = sinf(th);
        fwd = p ? -s : c;
        float dbl = (ky == 0) ? 1.f : 2.f;
        inv = (dbl / 1024.f) * (p ? -s : c);
    }
    Ff[(size_t)n * 128 + col]  = (__bf16)fwd;
    Bi[(size_t)col * 1024 + n] = (__bf16)inv;
}

// ======================= LayerNorm / eltwise kernels =======================
__global__ __launch_bounds__(256)
void k_ln1(const float* __restrict__ x, const float* __restrict__ g,
           const float* __restrict__ b, float* __restrict__ t,
           __bf16* __restrict__ tA)
{
    int token = blockIdx.x, bi = token >> 10, n = token & 1023;
    __shared__ float s1[256], s2[256];
    float vals[3], ls = 0.f, lq = 0.f;
    #pragma unroll
    for (int it = 0; it < 3; ++it) {
        int c = it * 256 + threadIdx.x;
        float v = x[((size_t)bi * 768 + c) * 1024 + n];
        vals[it] = v; ls += v; lq += v * v;
    }
    s1[threadIdx.x] = ls; s2[threadIdx.x] = lq; __syncthreads();
    for (int st = 128; st > 0; st >>= 1) {
        if (threadIdx.x < st) { s1[threadIdx.x] += s1[threadIdx.x + st];
                                s2[threadIdx.x] += s2[threadIdx.x + st]; }
        __syncthreads();
    }
    float mean = s1[0] / 768.f;
    float var  = s2[0] / 768.f - mean * mean;
    float is   = rsqrtf(var + 1e-5f);
    #pragma unroll
    for (int it = 0; it < 3; ++it) {
        int c = it * 256 + threadIdx.x;
        float y = (vals[it] - mean) * is * g[c] + b[c];
        t[(size_t)token * 768 + c] = y;
        tA[((size_t)bi * 768 + c) * 1024 + n] = (__bf16)y;   // [B*C, N] image layout
    }
}

__global__ __launch_bounds__(256)
void k_ln768(const float* __restrict__ in, const float* __restrict__ g,
             const float* __restrict__ b, __bf16* __restrict__ ob)
{
    int token = blockIdx.x;
    __shared__ float s1[256], s2[256];
    float vals[3], ls = 0.f, lq = 0.f;
    #pragma unroll
    for (int it = 0; it < 3; ++it) {
        int c = it * 256 + threadIdx.x;
        float v = in[(size_t)token * 768 + c];
        vals[it] = v; ls += v; lq += v * v;
    }
    s1[threadIdx.x] = ls; s2[threadIdx.x] = lq; __syncthreads();
    for (int st = 128; st > 0; st >>= 1) {
        if (threadIdx.x < st) { s1[threadIdx.x] += s1[threadIdx.x + st];
                                s2[threadIdx.x] += s2[threadIdx.x + st]; }
        __syncthreads();
    }
    float mean = s1[0] / 768.f, var = s2[0] / 768.f - mean * mean;
    float is = rsqrtf(var + 1e-5f);
    #pragma unroll
    for (int it = 0; it < 3; ++it) {
        int c = it * 256 + threadIdx.x;
        ob[(size_t)token * 768 + c] = (__bf16)((vals[it] - mean) * is * g[c] + b[c]);
    }
}

// LN(3072) + exact GELU, transpose-write to image layout [(b*3072+c)][n]
__global__ __launch_bounds__(256)
void k_ln_gelu_t(const float* __restrict__ u, const float* __restrict__ g,
                 const float* __restrict__ b, float* __restrict__ out)
{
    int token = blockIdx.x, bi = token >> 10, n = token & 1023;
    __shared__ float s1[256], s2[256];
    float vals[12], ls = 0.f, lq = 0.f;
    #pragma unroll
    for (int it = 0; it < 12; ++it) {
        int c = it * 256 + threadIdx.x;
        float v = u[(size_t)token * 3072 + c];
        vals[it] = v; ls += v; lq += v * v;
    }
    s1[threadIdx.x] = ls; s2[threadIdx.x] = lq; __syncthreads();
    for (int st = 128; st > 0; st >>= 1) {
        if (threadIdx.x < st) { s1[threadIdx.x] += s1[threadIdx.x + st];
                                s2[threadIdx.x] += s2[threadIdx.x + st]; }
        __syncthreads();
    }
    float mean = s1[0] / 3072.f, var = s2[0] / 3072.f - mean * mean;
    float is = rsqrtf(var + 1e-5f);
    #pragma unroll
    for (int it = 0; it < 12; ++it) {
        int c = it * 256 + threadIdx.x;
        float y = (vals[it] - mean) * is * g[c] + b[c];
        y = 0.5f * y * (1.f + erff(y * 0.7071067811865475f));
        out[((size_t)bi * 3072 + c) * 1024 + n] = y;
    }
}

// final LN(768) + residual + transpose to [B,C,H,W]
__global__ __launch_bounds__(256)
void k_ln_out(const float* __restrict__ u2, const float* __restrict__ g,
              const float* __restrict__ b, const float* __restrict__ t2,
              float* __restrict__ out)
{
    int token = blockIdx.x, bi = token >> 10, n = token & 1023;
    __shared__ float s1[256], s2[256];
    float vals[3], ls = 0.f, lq = 0.f;
    #pragma unroll
    for (int it = 0; it < 3; ++it) {
        int c = it * 256 + threadIdx.x;
        float v = u2[(size_t)token * 768 + c];
        vals[it] = v; ls += v; lq += v * v;
    }
    s1[threadIdx.x] = ls; s2[threadIdx.x] = lq; __syncthreads();
    for (int st = 128; st > 0; st >>= 1) {
        if (threadIdx.x < st) { s1[threadIdx.x] += s1[threadIdx.x + st];
                                s2[threadIdx.x] += s2[threadIdx.x + st]; }
        __syncthreads();
    }
    float mean = s1[0] / 768.f, var = s2[0] / 768.f - mean * mean;
    float is = rsqrtf(var + 1e-5f);
    #pragma unroll
    for (int it = 0; it < 3; ++it) {
        int c = it * 256 + threadIdx.x;
        float y = (vals[it] - mean) * is * g[c] + b[c];
        out[((size_t)bi * 768 + c) * 1024 + n] = t2[(size_t)token * 768 + c] + y;
    }
}

// softmax of k over the 1024 tokens, per (b,channel); writes ksᵀ in
// kv-GEMM A layout: KS[b][h][ch][n]
__global__ __launch_bounds__(256)
void k_softmax_k(const float* __restrict__ qkv, __bf16* __restrict__ KS)
{
    int bc = blockIdx.x;                 // b*768+c
    int bi = bc / 768, c = bc % 768;
    int h = c / 96, ch = c % 96;
    __shared__ float sm[256];
    const float* base = qkv + (size_t)bi * 1024 * 2304 + 768 + c;
    float vals[4], mx = -3.4e38f;
    #pragma unroll
    for (int it = 0; it < 4; ++it) {
        float v = base[(size_t)(it * 256 + threadIdx.x) * 2304];
        vals[it] = v; mx = fmaxf(mx, v);
    }
    sm[threadIdx.x] = mx; __syncthreads();
    for (int st = 128; st > 0; st >>= 1) {
        if (threadIdx.x < st) sm[threadIdx.x] = fmaxf(sm[threadIdx.x], sm[threadIdx.x + st]);
        __syncthreads();
    }
    mx = sm[0]; __syncthreads();
    float ls = 0.f;
    #pragma unroll
    for (int it = 0; it < 4; ++it) { vals[it] = expf(vals[it] - mx); ls += vals[it]; }
    sm[threadIdx.x] = ls; __syncthreads();
    for (int st = 128; st > 0; st >>= 1) {
        if (threadIdx.x < st) sm[threadIdx.x] += sm[threadIdx.x + st];
        __syncthreads();
    }
    float inv = 1.f / sm[0];
    __bf16* dst = KS + ((size_t)(bi * 8 + h) * 96 + ch) * 1024;
    #pragma unroll
    for (int it = 0; it < 4; ++it) {
        int n = it * 256 + threadIdx.x;
        dst[n] = (__bf16)(vals[it] * inv);
    }
}

// extract q,v in per-head layouts + v in image layout for spec_conv2
__global__ void k_qv_prep(const float* __restrict__ qkv, __bf16* __restrict__ Qb,
                          __bf16* __restrict__ Vb, __bf16* __restrict__ VA)
{
    size_t idx = (size_t)blockIdx.x * 256 + threadIdx.x;
    if (idx >= (size_t)16384 * 768) return;
    int c = (int)(idx % 768);
    size_t token = idx / 768;
    int bi = (int)(token >> 10), n = (int)(token & 1023);
    int h = c / 96, ch = c % 96;
    float q = qkv[token * 2304 + c];
    float v = qkv[token * 2304 + 1536 + c];
    size_t bh = (size_t)(bi * 8 + h);
    Qb[(bh * 1024 + n) * 96 + ch] = (__bf16)q;
    Vb[(bh * 1024 + n) * 96 + ch] = (__bf16)v;
    VA[((size_t)bi * 768 + c) * 1024 + n] = (__bf16)v;
}

// XF [12288][128] f32 -> XT [56][16][1536] bf16 (A operand of mode mixing)
__global__ void k_xt(const float* __restrict__ XF, __bf16* __restrict__ XT)
{
    size_t idx = (size_t)blockIdx.x * 256 + threadIdx.x;
    if (idx >= (size_t)56 * 16 * 1536) return;
    int k = (int)(idx % 1536);
    size_t r = idx / 1536;
    int b = (int)(r % 16), m = (int)(r / 16);
    int p = (k >= 768);
    int i = p ? (k - 768) : k;
    XT[idx] = (__bf16)XF[((size_t)b * 768 + i) * 128 + 2 * m + p];
}

// f [B*C,N] f32 -> token-major bf16 [B*N, C] (qkv GEMM A)
__global__ void k_tok_tr(const float* __restrict__ f, __bf16* __restrict__ fA)
{
    size_t idx = (size_t)blockIdx.x * 256 + threadIdx.x;
    if (idx >= (size_t)16384 * 768) return;
    int c = (int)(idx % 768);
    size_t token = idx / 768;
    int bi = (int)(token >> 10), n = (int)(token & 1023);
    fA[idx] = (__bf16)f[((size_t)bi * 768 + c) * 1024 + n];
}

// a = scale*fa + q*sigmoid(cv), bf16 token-major (proj GEMM A)
__global__ void k_combine(const float* __restrict__ qkv, const float* __restrict__ fa,
                          const float* __restrict__ cv, __bf16* __restrict__ Ap)
{
    size_t idx = (size_t)blockIdx.x * 256 + threadIdx.x;
    if (idx >= (size_t)16384 * 768) return;
    int c = (int)(idx % 768);
    size_t token = idx / 768;
    int bi = (int)(token >> 10), n = (int)(token & 1023);
    int h = c / 96, ch = c % 96;
    float q   = qkv[token * 2304 + c];
    float fav = fa[(((size_t)(bi * 8 + h)) * 1024 + n) * 96 + ch];
    float cvv = cv[((size_t)bi * 768 + c) * 1024 + n];
    float val = 0.10206207261596575f * fav + q * (1.f / (1.f + expf(-cvv)));
    Ap[idx] = (__bf16)val;
}

// depthwise 3x3 SAME + bias; transpose-write bf16 token-major (fc2 GEMM A)
__global__ __launch_bounds__(256)
void k_dwconv(const float* __restrict__ in, const float* __restrict__ w,
              const float* __restrict__ bias, __bf16* __restrict__ out)
{
    int plane = blockIdx.x;              // b*3072+cc
    int cc = plane % 3072, bi = plane / 3072;
    __shared__ float tile[32][33];
    const float* ip = in + (size_t)plane * 1024;
    #pragma unroll
    for (int e = 0; e < 4; ++e) {
        int idx = e * 256 + threadIdx.x;
        tile[idx >> 5][idx & 31] = ip[idx];
    }
    __syncthreads();
    float wg[9];
    #pragma unroll
    for (int j = 0; j < 9; ++j) wg[j] = w[cc * 9 + j];
    float bv = bias[cc];
    #pragma unroll
    for (int e = 0; e < 4; ++e) {
        int idx = e * 256 + threadIdx.x;
        int hh = idx >> 5, ww = idx & 31;
        float s = bv;
        #pragma unroll
        for (int dy = 0; dy < 3; ++dy) {
            int y = hh + dy - 1;
            if (y < 0 || y > 31) continue;
            #pragma unroll
            for (int dx = 0; dx < 3; ++dx) {
                int xw = ww + dx - 1;
                if (xw < 0 || xw > 31) continue;
                s += tile[y][xw] * wg[dy * 3 + dx];
            }
        }
        out[((size_t)bi * 1024 + idx) * 3072 + cc] = (__bf16)s;
    }
}

__global__ void k_f2bf(const float* __restrict__ in, __bf16* __restrict__ out, long long n)
{
    size_t i = (size_t)blockIdx.x * 256 + threadIdx.x;
    if (i < (size_t)n) out[i] = (__bf16)in[i];
}

__global__ void k_zero_bf(__bf16* __restrict__ p, long long n)
{
    size_t i = (size_t)blockIdx.x * 256 + threadIdx.x;
    if (i < (size_t)n) p[i] = (__bf16)0.f;
}

// ============================== launcher ===================================
extern "C" void kernel_launch(void* const* d_in, const int* in_sizes, int n_in,
                              void* d_out, int out_size, void* d_ws, size_t ws_size,
                              hipStream_t stream)
{
    (void)in_sizes; (void)n_in; (void)out_size; (void)ws_size;
    // dict order from setup_inputs()
    const float* x     = (const float*)d_in[0];
    const float* f1w1r = (const float*)d_in[1];
    const float* f1w1i = (const float*)d_in[2];
    const float* f1w2r = (const float*)d_in[3];
    const float* f1w2i = (const float*)d_in[4];
    const float* f2w1r = (const float*)d_in[5];
    const float* f2w1i = (const float*)d_in[6];
    const float* f2w2r = (const float*)d_in[7];
    const float* f2w2i = (const float*)d_in[8];
    const float* wqkv  = (const float*)d_in[9];
    const float* wproj = (const float*)d_in[10];
    const float* bproj = (const float*)d_in[11];
    const float* fc1w  = (const float*)d_in[12];
    const float* fc1b_ = (const float*)d_in[13];
    const float* fc2w  = (const float*)d_in[14];
    const float* fc2b_ = (const float*)d_in[15];
    const float* dww   = (const float*)d_in[16];
    const float* dwb   = (const float*)d_in[17];
    const float* ln1g  = (const float*)d_in[18];
    const float* ln1b  = (const float*)d_in[19];
    const float* ln2g  = (const float*)d_in[20];
    const float* ln2b  = (const float*)d_in[21];
    const float* mln1g = (const float*)d_in[22];
    const float* mln1b = (const float*)d_in[23];
    const float* mln2g = (const float*)d_in[24];
    const float* mln2b = (const float*)d_in[25];
    float* out = (float*)d_out;

    // bump allocator over d_ws (~1.1 GB total); 4 KB slack per buffer so the
    // guardless 128-wide tile staging may over-READ safely (never stored).
    char* wsp = (char*)d_ws;
    size_t off = 0;
    auto alloc = [&](size_t bytes) -> void* {
        void* p = wsp + off;
        off += ((bytes + 255) & ~(size_t)255) + 4096;
        return p;
    };
    __bf16* Ffwd  = (__bf16*)alloc((size_t)1024 * 128 * 2);
    __bf16* Binv  = (__bf16*)alloc((size_t)128 * 1024 * 2);
    __bf16* wqkvb = (__bf16*)alloc((size_t)768 * 2304 * 2);
    __bf16* wprjb = (__bf16*)alloc((size_t)768 * 768 * 2);
    __bf16* fc1bw = (__bf16*)alloc((size_t)768 * 3072 * 2);
    __bf16* fc2bw = (__bf16*)alloc((size_t)3072 * 768 * 2);
    float*  t     = (float*) alloc((size_t)16384 * 768 * 4);
    __bf16* tA    = (__bf16*)alloc((size_t)12288 * 1024 * 2);
    float*  XF    = (float*) alloc((size_t)12288 * 128 * 4);
    __bf16* XT    = (__bf16*)alloc((size_t)56 * 16 * 1536 * 2);
    __bf16* A3    = (__bf16*)alloc((size_t)12288 * 128 * 2);
    float*  fbuf  = (float*) alloc((size_t)12288 * 1024 * 4);  // f, later reused for cv
    __bf16* fA    = (__bf16*)alloc((size_t)16384 * 768 * 2);
    float*  qkv   = (float*) alloc((size_t)16384 * 2304 * 4);
    __bf16* KS    = (__bf16*)alloc((size_t)128 * 96 * 1024 * 2 + 32 * 1024 * 2);
    __bf16* Qb    = (__bf16*)alloc((size_t)128 * 1024 * 96 * 2);
    __bf16* Vb    = (__bf16*)alloc((size_t)128 * 1024 * 96 * 2);
    __bf16* VA    = (__bf16*)alloc((size_t)12288 * 1024 * 2);
    float*  kvf   = (float*) alloc((size_t)128 * 96 * 96 * 4);
    __bf16* kvb   = (__bf16*)alloc((size_t)128 * 96 * 96 * 2 + 32 * 96 * 2);
    float*  fab   = (float*) alloc((size_t)128 * 1024 * 96 * 4);
    __bf16* Aproj = (__bf16*)alloc((size_t)16384 * 768 * 2);
    float*  t2    = (float*) alloc((size_t)16384 * 768 * 4);
    __bf16* ubf   = (__bf16*)alloc((size_t)16384 * 768 * 2);
    float*  u1    = (float*) alloc((size_t)16384 * 3072 * 4);
    float*  uig   = (float*) alloc((size_t)16384 * 3072 * 4);
    __bf16* Afc2  = (__bf16*)alloc((size_t)16384 * 3072 * 2);
    float*  u2    = (float*) alloc((size_t)16384 * 768 * 4);

    auto gemm = [&](const __bf16* A, const __bf16* Bm, float* C, int M, int N, int K,
                    int lda, int ldb, int ldc, int batch,
                    long long sA, long long sB, long long sC,
                    const float* bias, const float* resid) {
        dim3 g((N + 127) / 128, (M + 127) / 128, batch);
        k_gemm<<<g, 256, 0, stream>>>(A, Bm, C, M, N, K, lda, ldb, ldc, sA, sB, sC, bias, resid);
    };
    auto elt = [&](size_t n) { return dim3((unsigned)((n + 255) / 256)); };

    // 0) basis tables + weight conversions + A3 padding zeros
    k_basis<<<1024, 128, 0, stream>>>(Ffwd, Binv);
    k_f2bf<<<elt((size_t)768 * 2304), 256, 0, stream>>>(wqkv, wqkvb, 768LL * 2304);
    k_f2bf<<<elt((size_t)768 * 768),  256, 0, stream>>>(wproj, wprjb, 768LL * 768);
    k_f2bf<<<elt((size_t)768 * 3072), 256, 0, stream>>>(fc1w, fc1bw, 768LL * 3072);
    k_f2bf<<<elt((size_t)3072 * 768), 256, 0, stream>>>(fc2w, fc2bw, 3072LL * 768);
    k_zero_bf<<<elt((size_t)12288 * 128), 256, 0, stream>>>(A3, 12288LL * 128);

    // 1) LN1 (also produces image-layout bf16 for the spectral conv)
    k_ln1<<<16384, 256, 0, stream>>>(x, ln1g, ln1b, t, tA);

    // 2) spectral conv #1: fwd DFT -> mode mix -> inv DFT
    gemm(tA, Ffwd, XF, 12288, 128, 1024, 1024, 128, 128, 1, 0, 0, 0, nullptr, nullptr);
    k_xt<<<elt((size_t)56 * 16 * 1536), 256, 0, stream>>>(XF, XT);
    k_modemix<<<dim3(12, 56), 256, 0, stream>>>(XT, f1w1r, f1w1i, f1w2r, f1w2i, A3);
    gemm(A3, Binv, fbuf, 12288, 1024, 128, 128, 1024, 1024, 1, 0, 0, 0, nullptr, nullptr);

    // 3) qkv projection
    k_tok_tr<<<elt((size_t)16384 * 768), 256, 0, stream>>>(fbuf, fA);
    gemm(fA, wqkvb, qkv, 16384, 2304, 768, 768, 2304, 2304, 1, 0, 0, 0, nullptr, nullptr);

    // 4) factorized attention
    k_softmax_k<<<12288, 256, 0, stream>>>(qkv, KS);
    k_qv_prep<<<elt((size_t)16384 * 768), 256, 0, stream>>>(qkv, Qb, Vb, VA);
    gemm(KS, Vb, kvf, 96, 96, 1024, 1024, 96, 96, 128,
         96LL * 1024, 1024LL * 96, 96LL * 96, nullptr, nullptr);
    k_f2bf<<<elt((size_t)128 * 96 * 96), 256, 0, stream>>>(kvf, kvb, 128LL * 96 * 96);
    gemm(Qb, kvb, fab, 1024, 96, 96, 96, 96, 96, 128,
         1024LL * 96, 96LL * 96, 1024LL * 96, nullptr, nullptr);

    // 5) spectral conv #2 on v (reuses XF/XT/A3; cv lands in fbuf)
    gemm(VA, Ffwd, XF, 12288, 128, 1024, 1024, 128, 128, 1, 0, 0, 0, nullptr, nullptr);
    k_xt<<<elt((size_t)56 * 16 * 1536), 256, 0, stream>>>(XF, XT);
    k_modemix<<<dim3(12, 56), 256, 0, stream>>>(XT, f2w1r, f2w1i, f2w2r, f2w2i, A3);
    gemm(A3, Binv, fbuf, 12288, 1024, 128, 128, 1024, 1024, 1, 0, 0, 0, nullptr, nullptr);

    // 6) combine + output projection (+bias +residual post-LN t)
    k_combine<<<elt((size_t)16384 * 768), 256, 0, stream>>>(qkv, fab, fbuf, Aproj);
    gemm(Aproj, wprjb, t2, 16384, 768, 768, 768, 768, 768, 1, 0, 0, 0, bproj, t);

    // 7) MLP
    k_ln768<<<16384, 256, 0, stream>>>(t2, ln2g, ln2b, ubf);
    gemm(ubf, fc1bw, u1, 16384, 3072, 768, 768, 3072, 3072, 1, 0, 0, 0, fc1b_, nullptr);
    k_ln_gelu_t<<<16384, 256, 0, stream>>>(u1, mln1g, mln1b, uig);
    k_dwconv<<<49152, 256, 0, stream>>>(uig, dww, dwb, Afc2);
    gemm(Afc2, fc2bw, u2, 16384, 768, 3072, 3072, 768, 768, 1, 0, 0, 0, fc2b_, nullptr);
    k_ln_out<<<16384, 256, 0, stream>>>(u2, mln2g, mln2b, t2, out);
}
